// Decoder_33964601376783
// MI455X (gfx1250) — matrix operand order repaired and checked by reading, not compiled, not run on hardware
//
#include <hip/hip_runtime.h>
#include <hip/hip_bf16.h>

// ---------------------------------------------------------------------------
// Transformer decoder (reference bug preserved: only layer L-1 matters, since
// every layer reads the ORIGINAL decoderInput and only `last` is returned).
// All GEMMs + attention run on v_wmma_f32_16x16x32_f16 (f32 accumulate).
// ---------------------------------------------------------------------------

typedef __attribute__((ext_vector_type(16))) _Float16 v16h;
typedef __attribute__((ext_vector_type(8)))  _Float16 v8h;
typedef __attribute__((ext_vector_type(8)))  float    v8f;

#define D_MODEL 1024
#define NHEAD   8
#define DKV     128
#define SEQ     2048
#define FFDIM   4096
#define NLAYER  6

// Build a 16-half fragment from two 16-byte LDS/global chunks.
__device__ inline v16h ld_frag_pair(const _Float16* p0, const _Float16* p1) {
  v8h lo = *(const v8h*)p0;
  v8h hi = *(const v8h*)p1;
  v16h r;
#pragma unroll
  for (int i = 0; i < 8; ++i) { r[i] = lo[i]; r[i + 8] = hi[i]; }
  return r;
}

__device__ inline v8f wmma_f16(v16h a, v16h b, v8f c) {
  // D = A(16x32,f16) * B(32x16,f16) + C(16x16,f32)
  return __builtin_amdgcn_wmma_f32_16x16x32_f16(
      /*neg_a=*/false, a, /*neg_b=*/false, b,
      /*c_mod=*/(short)0, c, /*reuse_a=*/false, /*reuse_b=*/false);
}

// Reduce across the 16 lanes of a half-wave (lanes 0-15 and 16-31 separately).
__device__ inline float red16_max(float v) {
  v = fmaxf(v, __shfl_xor(v, 1, 32));
  v = fmaxf(v, __shfl_xor(v, 2, 32));
  v = fmaxf(v, __shfl_xor(v, 4, 32));
  v = fmaxf(v, __shfl_xor(v, 8, 32));
  return v;
}
__device__ inline float red16_sum(float v) {
  v += __shfl_xor(v, 1, 32);
  v += __shfl_xor(v, 2, 32);
  v += __shfl_xor(v, 4, 32);
  v += __shfl_xor(v, 8, 32);
  return v;
}

// Stage 16 contiguous elements into LDS as f16 (fp32 and f16 sources).
__device__ inline void stage16(const float* p, _Float16* dst) {
#pragma unroll
  for (int i4 = 0; i4 < 4; ++i4) {
    float4 f = *(const float4*)(p + i4 * 4);
    dst[i4 * 4 + 0] = (_Float16)f.x;
    dst[i4 * 4 + 1] = (_Float16)f.y;
    dst[i4 * 4 + 2] = (_Float16)f.z;
    dst[i4 * 4 + 3] = (_Float16)f.w;
  }
}
__device__ inline void stage16(const _Float16* p, _Float16* dst) {
#pragma unroll
  for (int i = 0; i < 16; i += 8) {
    v8h v = *(const v8h*)(p + i);
#pragma unroll
    for (int e = 0; e < 8; ++e) dst[i + e] = v[e];
  }
}

// ---------------------------------------------------------------------------
// Tiled WMMA GEMM: C[z] = A * B[z] + bias[z], optional ReLU.
// Block: 128 threads (4 waves). Tile 128(M) x 64(N), K-step 32.
// Each wave computes a 32x64 sub-tile: 2 A-frags x 4 B-frags = 8 WMMA/K-step
// (32 WMMA per block per K-step; B fragments amortized over 32 rows).
// All dimensions are multiples of 128/64/32 -> no bounds checks needed.
// ---------------------------------------------------------------------------
template <typename AT, typename OT, bool RELU>
__global__ __launch_bounds__(128) void gemm_wmma(
    const AT* __restrict__ A, int lda,
    const float* __restrict__ B, int ldb, long bStrideZ,
    const float* __restrict__ bias, int biasStrideZ,
    OT* __restrict__ C, int ldc, long cStrideZ, int K) {
  __shared__ __align__(32) _Float16 As[128][40];  // [m][k], padded rows (80B)
  __shared__ __align__(32) _Float16 Bs[64][40];   // [n][k] (transposed), padded

  const int z = blockIdx.z;
  B    += (size_t)z * bStrideZ;
  bias += (size_t)z * biasStrideZ;
  C    += (size_t)z * cStrideZ;

  const int m0 = blockIdx.x * 128;
  const int n0 = blockIdx.y * 64;
  const int t    = threadIdx.x;
  const int lane = t & 31;
  const int wv   = t >> 5;
  const int ln15 = lane & 15;
  const int kg   = lane >> 4;

  const int bk = t >> 2, bn = (t & 3) * 16;  // B staging: 32 k x 64 n

  v8f zf = {};
  v8f acc[2][4];
#pragma unroll
  for (int mi = 0; mi < 2; ++mi)
#pragma unroll
    for (int j = 0; j < 4; ++j) acc[mi][j] = zf;

  for (int k0 = 0; k0 < K; k0 += 32) {
    // stage A: thread t owns row t (32 k-values, 128B contiguous read)
    {
      const AT* ap = A + (size_t)(m0 + t) * lda + k0;
      stage16(ap, &As[t][0]);
      stage16(ap + 16, &As[t][16]);
    }
    // stage B transposed
    {
      const float* bp = B + (size_t)(k0 + bk) * ldb + n0 + bn;
#pragma unroll
      for (int i4 = 0; i4 < 4; ++i4) {
        float4 f = *(const float4*)(bp + i4 * 4);
        Bs[bn + i4 * 4 + 0][bk] = (_Float16)f.x;
        Bs[bn + i4 * 4 + 1][bk] = (_Float16)f.y;
        Bs[bn + i4 * 4 + 2][bk] = (_Float16)f.z;
        Bs[bn + i4 * 4 + 3][bk] = (_Float16)f.w;
      }
    }
    __syncthreads();

    // A fragments: elems 0-7 -> k = 8*kg + j ; elems 8-15 -> k = 16 + 8*kg + j
    v16h a[2];
#pragma unroll
    for (int mi = 0; mi < 2; ++mi) {
      const int am = wv * 32 + mi * 16 + ln15;
      a[mi] = ld_frag_pair(&As[am][8 * kg], &As[am][16 + 8 * kg]);
    }
#pragma unroll
    for (int j = 0; j < 4; ++j) {
      const int bn2 = j * 16 + ln15;
      // B fragment: elems 0-7 -> k = 8*kg + j ; elems 8-15 -> k = 16 + 8*kg + j
      v16h b = ld_frag_pair(&Bs[bn2][8 * kg], &Bs[bn2][16 + 8 * kg]);
#pragma unroll
      for (int mi = 0; mi < 2; ++mi) acc[mi][j] = wmma_f16(a[mi], b, acc[mi][j]);
    }
    __syncthreads();
  }

  // Epilogue: D layout -> row = r + 8*(lane/16), col = lane%16 per 16x16 tile.
#pragma unroll
  for (int mi = 0; mi < 2; ++mi) {
#pragma unroll
    for (int j = 0; j < 4; ++j) {
      const int col = n0 + j * 16 + ln15;
      const float bb = bias[col];
#pragma unroll
      for (int r = 0; r < 8; ++r) {
        const int row = m0 + wv * 32 + mi * 16 + r + 8 * kg;
        float v = acc[mi][j][r] + bb;
        if (RELU) v = fmaxf(v, 0.0f);
        C[(size_t)row * ldc + col] = (OT)v;
      }
    }
  }
}

// ---------------------------------------------------------------------------
// Flash attention: 4 waves per block, each owning a 16-row query tile of the
// SAME head, sharing one staged 32x128 V chunk per iteration (4x less V
// traffic). f16 Q/K/V in head-major [H][T][128] layout; online softmax with
// 1/D scaling (reference divides by D, not sqrt(D)). Output layout
// out[s][h*128 + v] (concat heads). Causal waves see some fully-masked tail
// chunks: chunk-max=-inf => scale=1, p=0, numerically safe.
// ---------------------------------------------------------------------------
__global__ __launch_bounds__(128) void attn_wmma(
    const _Float16* __restrict__ Q, const _Float16* __restrict__ Kh_,
    const _Float16* __restrict__ Vh_, float* __restrict__ out, int T,
    int causal) {
  __shared__ __align__(32) _Float16 Ps[4][16][32];  // per-wave prob tiles
  __shared__ __align__(32) _Float16 Vt[128][40];    // V chunk transposed [v][t]

  const int h    = blockIdx.y;
  const int t_   = threadIdx.x;
  const int lane = t_ & 31;
  const int wv   = t_ >> 5;
  const int ln15 = lane & 15;
  const int kg   = lane >> 4;
  const int s0   = blockIdx.x * 64 + wv * 16;  // this wave's query rows

  const _Float16* Qh = Q   + (size_t)h * SEQ * DKV;
  const _Float16* Kh = Kh_ + (size_t)h * T * DKV;
  const _Float16* Vh = Vh_ + (size_t)h * T * DKV;

  // Q fragments for the 16-row tile: DK=128 -> 4 WMMA K-chunks, kept in VGPRs.
  v16h qf[4];
#pragma unroll
  for (int kc = 0; kc < 4; ++kc) {
    const _Float16* qp = Qh + (size_t)(s0 + ln15) * DKV + kc * 32 + 8 * kg;
    qf[kc] = ld_frag_pair(qp, qp + 16);
  }

  v8f zf = {};
  v8f acc[8];  // 16 x 128 f32 output accumulator
#pragma unroll
  for (int j = 0; j < 8; ++j) acc[j] = zf;
  float rmax[8], rsum[8];
#pragma unroll
  for (int r = 0; r < 8; ++r) { rmax[r] = -__builtin_inff(); rsum[r] = 0.f; }

  const float invD = 1.0f / (float)D_MODEL;
  // Uniform loop bound across the block (max over its 4 wave tiles).
  const int tEnd = causal ? (blockIdx.x * 64 + 64) : T;

  // V staging assignment: thread owns row (t_>>2), 32 cols at ((t_&3)*32).
  const int vr  = t_ >> 2;
  const int vc0 = (t_ & 3) * 32;

  for (int t0 = 0; t0 < tEnd; t0 += 32) {
    // Stage V chunk transposed once per block: Vt[v][t] = Vh[t0+t][v].
    {
      const _Float16* vp = Vh + (size_t)(t0 + vr) * DKV + vc0;
#pragma unroll
      for (int cc = 0; cc < 4; ++cc) {
        v8h vv = *(const v8h*)(vp + cc * 8);
#pragma unroll
        for (int e = 0; e < 8; ++e) Vt[vc0 + cc * 8 + e][vr] = vv[e];
      }
    }

    // scores = Q * K^T for two 16x16 column tiles (t0..t0+31)
    v8f sc[2];
#pragma unroll
    for (int half = 0; half < 2; ++half) {
      v8f s = zf;
#pragma unroll
      for (int kc = 0; kc < 4; ++kc) {
        // B fragment of K^T: col n = t-index, elems 0-7 -> dk = kc*32 + 8*kg + j, elems 8-15 -> +16
        const _Float16* kp =
            Kh + (size_t)(t0 + half * 16 + ln15) * DKV + kc * 32 + 8 * kg;
        v16h kf = ld_frag_pair(kp, kp + 16);
        s = wmma_f16(qf[kc], kf, s);
      }
      sc[half] = s;
    }

    // Online softmax (per row r; lanes 0-15 own rows 0-7, lanes 16-31 rows 8-15)
#pragma unroll
    for (int r = 0; r < 8; ++r) {
      const int row  = s0 + r + 8 * kg;
      const int colA = t0 + ln15;
      const int colB = t0 + 16 + ln15;
      float vA = sc[0][r] * invD;
      float vB = sc[1][r] * invD;
      if (causal && colA > row) vA = -__builtin_inff();
      if (causal && colB > row) vB = -__builtin_inff();
      float cm   = red16_max(fmaxf(vA, vB));
      float mNew = fmaxf(rmax[r], cm);
      float scl  = __expf(rmax[r] - mNew);
      float pA   = __expf(vA - mNew);
      float pB   = __expf(vB - mNew);
      float psum = red16_sum(pA + pB);
      rsum[r] = rsum[r] * scl + psum;
      rmax[r] = mNew;
#pragma unroll
      for (int j = 0; j < 8; ++j) acc[j][r] *= scl;
      Ps[wv][r + 8 * kg][ln15]      = (_Float16)pA;
      Ps[wv][r + 8 * kg][16 + ln15] = (_Float16)pB;
    }
    __syncthreads();

    // acc += P(16x32) * V(32x128)
    v16h pf = ld_frag_pair(&Ps[wv][ln15][8 * kg], &Ps[wv][ln15][16 + 8 * kg]);
#pragma unroll
    for (int j = 0; j < 8; ++j) {
      const int vc = j * 16 + ln15;
      v16h bf = ld_frag_pair(&Vt[vc][8 * kg], &Vt[vc][16 + 8 * kg]);
      acc[j] = wmma_f16(pf, bf, acc[j]);
    }
    __syncthreads();
  }

  // Normalize and write concat-head output.
#pragma unroll
  for (int j = 0; j < 8; ++j) {
#pragma unroll
    for (int r = 0; r < 8; ++r) {
      const int row = s0 + r + 8 * kg;
      out[(size_t)row * (NHEAD * DKV) + h * DKV + j * 16 + ln15] =
          acc[j][r] / rsum[r];
    }
  }
}

// ---------------------------------------------------------------------------
// out = LayerNorm(xa + xb) * gamma + beta  (row-wise over D_MODEL=1024)
// ---------------------------------------------------------------------------
__global__ __launch_bounds__(256) void add_ln(
    const float* __restrict__ xa, const float* __restrict__ xb,
    const float* __restrict__ g, const float* __restrict__ b,
    float* __restrict__ out) {
  __shared__ float sred[8];
  const int row = blockIdx.x;
  const int t   = threadIdx.x;
  const size_t base = (size_t)row * D_MODEL;

  float v[4];
#pragma unroll
  for (int i = 0; i < 4; ++i) {
    const int c = t + i * 256;
    v[i] = xa[base + c] + xb[base + c];
  }
  float s = v[0] + v[1] + v[2] + v[3];
#pragma unroll
  for (int m = 1; m < 32; m <<= 1) s += __shfl_xor(s, m, 32);
  if ((t & 31) == 0) sred[t >> 5] = s;
  __syncthreads();
  float mu = 0.f;
#pragma unroll
  for (int i = 0; i < 8; ++i) mu += sred[i];
  mu *= (1.0f / D_MODEL);
  __syncthreads();

  float s2 = 0.f;
#pragma unroll
  for (int i = 0; i < 4; ++i) { float d = v[i] - mu; s2 += d * d; }
#pragma unroll
  for (int m = 1; m < 32; m <<= 1) s2 += __shfl_xor(s2, m, 32);
  if ((t & 31) == 0) sred[t >> 5] = s2;
  __syncthreads();
  float var = 0.f;
#pragma unroll
  for (int i = 0; i < 8; ++i) var += sred[i];
  var *= (1.0f / D_MODEL);
  const float inv = rsqrtf(var + 1e-5f);
#pragma unroll
  for (int i = 0; i < 4; ++i) {
    const int c = t + i * 256;
    out[base + c] = (v[i] - mu) * inv * g[c] + b[c];
  }
}

// ---------------------------------------------------------------------------
extern "C" void kernel_launch(void* const* d_in, const int* in_sizes, int n_in,
                              void* d_out, int out_size, void* d_ws,
                              size_t ws_size, hipStream_t stream) {
  (void)in_sizes; (void)n_in; (void)out_size; (void)ws_size;

  const float* x    = (const float*)d_in[0];
  const float* enc  = (const float*)d_in[1];
  const float* Wq1  = (const float*)d_in[2];
  const float* bq1  = (const float*)d_in[3];
  const float* Wk1  = (const float*)d_in[4];
  const float* bk1  = (const float*)d_in[5];
  const float* Wv1  = (const float*)d_in[6];
  const float* bv1  = (const float*)d_in[7];
  const float* Wq2  = (const float*)d_in[8];
  const float* bq2  = (const float*)d_in[9];
  const float* Wk2  = (const float*)d_in[10];
  const float* bk2  = (const float*)d_in[11];
  const float* Wv2  = (const float*)d_in[12];
  const float* bv2  = (const float*)d_in[13];
  const float* Wff1 = (const float*)d_in[14];
  const float* bff1 = (const float*)d_in[15];
  const float* Wff2 = (const float*)d_in[16];
  const float* bff2 = (const float*)d_in[17];
  const float* gmm  = (const float*)d_in[18];
  const float* bet  = (const float*)d_in[19];
  float* outp = (float*)d_out;

  // Only layer L-1 affects the result (reference re-reads decoderInput).
  const int l = NLAYER - 1;
  const size_t wO = (size_t)l * NHEAD * D_MODEL * DKV;
  const size_t bO = (size_t)l * NHEAD * DKV;

  char* ws = (char*)d_ws;
  _Float16* Qh = (_Float16*)ws; ws += (size_t)NHEAD * SEQ * DKV * 2;
  _Float16* Kh = (_Float16*)ws; ws += (size_t)NHEAD * SEQ * DKV * 2;
  _Float16* Vh = (_Float16*)ws; ws += (size_t)NHEAD * SEQ * DKV * 2;
  float* tmp   = (float*)ws;    ws += (size_t)SEQ * D_MODEL * 4;
  float* h1    = (float*)ws;    ws += (size_t)SEQ * D_MODEL * 4;
  float* h2    = (float*)ws;    ws += (size_t)SEQ * D_MODEL * 4;
  _Float16* ffm = (_Float16*)ws;

  const dim3 blk(128);
  const dim3 gProj(SEQ / 128, DKV / 64, NHEAD);  // per-head N=128 projections
  const dim3 gAttn(SEQ / 64, NHEAD);
  const long bSZ = (long)D_MODEL * DKV;
  const long cSZ = (long)SEQ * DKV;

  // --- self attention (reads ORIGINAL decoder input) ---
  gemm_wmma<float, _Float16, false><<<gProj, blk, 0, stream>>>(
      x, D_MODEL, Wq1 + wO, DKV, bSZ, bq1 + bO, DKV, Qh, DKV, cSZ, D_MODEL);
  gemm_wmma<float, _Float16, false><<<gProj, blk, 0, stream>>>(
      x, D_MODEL, Wk1 + wO, DKV, bSZ, bk1 + bO, DKV, Kh, DKV, cSZ, D_MODEL);
  gemm_wmma<float, _Float16, false><<<gProj, blk, 0, stream>>>(
      x, D_MODEL, Wv1 + wO, DKV, bSZ, bv1 + bO, DKV, Vh, DKV, cSZ, D_MODEL);
  attn_wmma<<<gAttn, blk, 0, stream>>>(Qh, Kh, Vh, tmp, SEQ, 1);
  add_ln<<<dim3(SEQ), dim3(256), 0, stream>>>(x, tmp, gmm, bet, h1);

  // --- cross attention (K/V from encoder output) ---
  gemm_wmma<float, _Float16, false><<<gProj, blk, 0, stream>>>(
      h1, D_MODEL, Wq2 + wO, DKV, bSZ, bq2 + bO, DKV, Qh, DKV, cSZ, D_MODEL);
  gemm_wmma<float, _Float16, false><<<gProj, blk, 0, stream>>>(
      enc, D_MODEL, Wk2 + wO, DKV, bSZ, bk2 + bO, DKV, Kh, DKV, cSZ, D_MODEL);
  gemm_wmma<float, _Float16, false><<<gProj, blk, 0, stream>>>(
      enc, D_MODEL, Wv2 + wO, DKV, bSZ, bv2 + bO, DKV, Vh, DKV, cSZ, D_MODEL);
  attn_wmma<<<gAttn, blk, 0, stream>>>(Qh, Kh, Vh, tmp, SEQ, 0);
  add_ln<<<dim3(SEQ), dim3(256), 0, stream>>>(h1, tmp, gmm, bet, h2);

  // --- FFN ---
  gemm_wmma<float, _Float16, true><<<dim3(SEQ / 128, FFDIM / 64, 1), blk, 0,
                                     stream>>>(
      h2, D_MODEL, Wff1 + (size_t)l * D_MODEL * FFDIM, FFDIM, 0,
      bff1 + (size_t)l * FFDIM, 0, ffm, FFDIM, 0, D_MODEL);
  gemm_wmma<_Float16, float, false><<<dim3(SEQ / 128, D_MODEL / 64, 1), blk, 0,
                                      stream>>>(
      ffm, FFDIM, Wff2 + (size_t)l * FFDIM * D_MODEL, D_MODEL, 0,
      bff2 + (size_t)l * D_MODEL, 0, tmp, D_MODEL, 0, FFDIM);
  add_ln<<<dim3(SEQ), dim3(256), 0, stream>>>(h2, tmp, gmm, bet, outp);
}
